// SwitchedConvHardRouting_83863531422097
// MI455X (gfx1250) — matrix operand order repaired and checked by reading, not compiled, not run on hardware
//
#include <hip/hip_runtime.h>
#include <hip/hip_bf16.h>

typedef __attribute__((ext_vector_type(16))) _Float16 v16h;
typedef __attribute__((ext_vector_type(8)))  _Float16 v8h;
typedef __attribute__((ext_vector_type(8)))  float    v8f;

#define BB   8
#define ICN  256
#define OCN  256
#define SSN  8
#define HH   48
#define WWD  48
#define NPIX (BB * HH * WWD)          // 18432
#define HWC  (HH * WWD)               // 2304

// ---------------- workspace layout (bytes) ----------------
// Apack f16 : SSN*9*OCN*ICN = 9,437,184 B
// xpack f16 : NPIX*ICN      = 9,437,184 B
// sel   f32 : NPIX*SSN      =   589,824 B
#define APACK_BYTES ((size_t)SSN * 9 * OCN * ICN * 2)
#define XPACK_BYTES ((size_t)NPIX * ICN * 2)

// ---- repack weight [OC][IC][S][3][3] f32 -> Apack[s][ky][kx][oc][ic] f16 ----
__global__ void pack_w_kernel(const float* __restrict__ w, _Float16* __restrict__ apack) {
    int i    = blockIdx.x * 256 + threadIdx.x;     // flat index in Apack order
    int ic   = i & 255;
    int oc   = (i >> 8) & 255;
    int stap = i >> 16;                            // 0..71 = s*9 + tap
    int s    = stap / 9;
    int tap  = stap % 9;
    int ky   = tap / 3, kx = tap % 3;
    float v = w[(((oc * ICN + ic) * SSN + s) * 3 + ky) * 3 + kx];
    apack[i] = (_Float16)v;
}

// ---- repack x NCHW f32 -> NHWC f16 ----
__global__ void pack_x_kernel(const float* __restrict__ x, _Float16* __restrict__ xp) {
    int i   = blockIdx.x * 256 + threadIdx.x;      // flat in xpack order
    int ic  = i & 255;
    int pix = i >> 8;
    int b   = pix / HWC;
    int r   = pix % HWC;
    xp[i] = (_Float16)x[(b * ICN + ic) * HWC + r];
}

// ---- coupler 1x1 conv + softmax -> sel[n][s] f32 ----
__global__ void coupler_kernel(const float* __restrict__ x, const float* __restrict__ cw,
                               const float* __restrict__ cb, float* __restrict__ sel) {
    int n = blockIdx.x * 256 + threadIdx.x;        // pixel index
    int b = n / HWC;
    int r = n % HWC;
    const float* xb = x + (size_t)b * ICN * HWC + r;
    float lg[SSN];
#pragma unroll
    for (int s = 0; s < SSN; ++s) lg[s] = cb[s];
    for (int ic = 0; ic < ICN; ++ic) {
        float xv = xb[(size_t)ic * HWC];
#pragma unroll
        for (int s = 0; s < SSN; ++s) lg[s] = fmaf(xv, cw[s * ICN + ic], lg[s]);
    }
    float m = lg[0];
#pragma unroll
    for (int s = 1; s < SSN; ++s) m = fmaxf(m, lg[s]);
    float e[SSN], sum = 0.0f;
#pragma unroll
    for (int s = 0; s < SSN; ++s) { e[s] = __expf(lg[s] - m); sum += e[s]; }
    float inv = 1.0f / sum;
#pragma unroll
    for (int s = 0; s < SSN; ++s) sel[n * SSN + s] = e[s] * inv;
}

__device__ inline v16h cat16(v8h lo, v8h hi) {
    v16h r;
#pragma unroll
    for (int i = 0; i < 8; ++i) { r[i] = lo[i]; r[i + 8] = hi[i]; }
    return r;
}

// ---- main switched-conv implicit GEMM ----
// grid = NPIX/32 blocks, 128 threads (4 waves). Wave wv: OC [wv*64, wv*64+64),
// block: pixels [blockIdx.x*32, +32) as two 16-pixel row-aligned tiles.
__global__ __launch_bounds__(128) void swconv_kernel(
        const _Float16* __restrict__ apack, const _Float16* __restrict__ xp,
        const float* __restrict__ sel, const float* __restrict__ bias,
        float* __restrict__ out) {
    const int lane   = threadIdx.x & 31;
    const int wv     = threadIdx.x >> 5;
    const int row    = lane & 15;          // A row (oc) / B col (pixel) / C col
    const int half   = lane >> 4;          // K half-select
    const int ocbase = wv * 64;
    const int t16    = blockIdx.x * 2;

    // lane-dependent, loop-invariant A offset (in halves): row*ICN (+ half*16 via ap[2*half])
    const int arow = row * ICN;

    int bj[2], hj[2], w0[2];
#pragma unroll
    for (int j = 0; j < 2; ++j) {
        int n16 = (t16 + j) * 16;
        bj[j] = n16 / HWC;
        int r = n16 % HWC;
        hj[j] = r / WWD;
        w0[j] = r % WWD;                   // 0, 16, or 32
    }

    // per-column selector scales (f16), hoisted for all taps/ic
    _Float16 sf[2][SSN];
#pragma unroll
    for (int j = 0; j < 2; ++j)
#pragma unroll
        for (int s = 0; s < SSN; ++s)
            sf[j][s] = (_Float16)sel[((t16 + j) * 16 + row) * SSN + s];

    v8f zero8 = {};
    v8f acc[4][2];
#pragma unroll
    for (int t = 0; t < 4; ++t)
#pragma unroll
        for (int j = 0; j < 2; ++j) acc[t][j] = zero8;

#pragma unroll 1
    for (int tap = 0; tap < 9; ++tap) {
        const int dy = tap / 3, dx = tap % 3;
        bool inr[2];
        int  voffx[2];                      // lane-dependent xpack offset (halves), per tap
#pragma unroll
        for (int j = 0; j < 2; ++j) {
            int ih = hj[j] + dy - 1;
            int iw = w0[j] + row + dx - 1;
            inr[j]   = (ih >= 0) && (ih < HH) && (iw >= 0) && (iw < WWD);
            voffx[j] = ((bj[j] * HH + ih) * WWD + iw) * ICN + 16 * half;
        }
#pragma unroll 1
        for (int ic0 = 0; ic0 < ICN; ic0 += 32) {
            v16h bx[2];
#pragma unroll
            for (int j = 0; j < 2; ++j) {
                v8h lo, hi;
                if (inr[j]) {
                    const v8h* p = (const v8h*)(xp + voffx[j] + ic0);
                    lo = p[0]; hi = p[1];
                } else {
                    lo = v8h{}; hi = v8h{};
                }
                bx[j] = cat16(lo, hi);
            }
#pragma unroll
            for (int s = 0; s < SSN; ++s) {
                v16h bs0, bs1;
                _Float16 f0 = sf[0][s], f1 = sf[1][s];
#pragma unroll
                for (int i = 0; i < 16; ++i) { bs0[i] = bx[0][i] * f0; bs1[i] = bx[1][i] * f1; }
#pragma unroll
                for (int t = 0; t < 4; ++t) {
                    // uniform scalar base: s/tap/t/ic0 components; lane part is arow (+16*half)
                    const _Float16* au =
                        apack + (size_t)((s * 9 + tap) * OCN + ocbase + t * 16) * ICN + ic0;
                    const v8h* ap = (const v8h*)(au + arow);
                    v16h a = cat16(ap[2 * half], ap[2 * half + 1]);
                    acc[t][0] = __builtin_amdgcn_wmma_f32_16x16x32_f16(
                        false, a, false, bs0, (short)0, acc[t][0], false, false);
                    acc[t][1] = __builtin_amdgcn_wmma_f32_16x16x32_f16(
                        false, a, false, bs1, (short)0, acc[t][1], false, false);
                }
            }
        }
    }

    // epilogue: C/D layout -> VGPR v holds M = v + 8*half, N = row; add bias
#pragma unroll
    for (int t = 0; t < 4; ++t)
#pragma unroll
        for (int j = 0; j < 2; ++j)
#pragma unroll
            for (int v = 0; v < 8; ++v) {
                int oc = ocbase + t * 16 + v + 8 * half;
                int o  = ((bj[j] * OCN + oc) * HH + hj[j]) * WWD + w0[j] + row;
                out[o] = acc[t][j][v] + bias[oc];
            }
}

extern "C" void kernel_launch(void* const* d_in, const int* in_sizes, int n_in,
                              void* d_out, int out_size, void* d_ws, size_t ws_size,
                              hipStream_t stream) {
    const float* x    = (const float*)d_in[0];
    const float* cw   = (const float*)d_in[1];
    const float* cb   = (const float*)d_in[2];
    const float* wt   = (const float*)d_in[3];
    const float* bias = (const float*)d_in[4];
    float* out = (float*)d_out;

    char* ws = (char*)d_ws;
    _Float16* apack = (_Float16*)ws;
    _Float16* xpack = (_Float16*)(ws + APACK_BYTES);
    float*    sel   = (float*)(ws + APACK_BYTES + XPACK_BYTES);

    pack_w_kernel<<<(SSN * 9 * OCN * ICN) / 256, 256, 0, stream>>>(wt, apack);
    pack_x_kernel<<<(NPIX * ICN) / 256, 256, 0, stream>>>(x, xpack);
    coupler_kernel<<<NPIX / 256, 256, 0, stream>>>(x, cw, cb, sel);
    swconv_kernel<<<NPIX / 32, 128, 0, stream>>>(apack, xpack, sel, bias, out);
}
